// CanonicalMicrocircuit_17274358464668
// MI455X (gfx1250) — compile-verified, matrix operand order checked by reading, not run on hardware
//
#include <hip/hip_runtime.h>
#include <hip/hip_bf16.h>

// Live computation of the reference (after dead-code elimination; see analysis):
//   x[c]  = relu((1-a_c)*(blat_e[c,0]+bfb_e[c,0]) - thr_c);  x[c] /= (||x[c]||+1e-8)
//   out[c]= relu(Wexc[c,0] @ x[c] + bexc[c,0])
//   h     = relu(Wg1 @ out.flat + bg1)
//   final = out.flat * (1 - sigmoid(Wg2 @ h + bg2))
// 96 MB of fp32 weight streaming, ~48 MFLOP -> HBM-bound (~4.1 us at 23.3 TB/s).
// Matvecs are executed with v_wmma_f32_16x16x32_f16 (f16 inputs, f32 accum).

#define C_  8
#define H_  1024
#define L_  4
#define CH_ 8192   // C_*H_

typedef __attribute__((ext_vector_type(16))) _Float16 v16h;
typedef __attribute__((ext_vector_type(8)))  _Float16 v8h;
typedef __attribute__((ext_vector_type(8)))  float    v8f;

// One wave computes 16 rows of W(16 x kLen) @ x, f32 accumulation via WMMA.
// A tile (16x32 f16): lane<16 holds row M=lane, K slots {k0..k0+7, k0+16..k0+23};
// lane>=16 holds row M=lane-16, K slots {k0+8..k0+15, k0+24..k0+31} (ISA 7.12.2).
// B tile is the vector broadcast across all 16 columns: every lane's K slots get
// x[k] at the mirrored K positions, so every D column equals the matvec result.
__device__ __forceinline__ v8f wmma_matvec_rows16(
    const float* __restrict__ Wrow0, int ldw,
    const _Float16* xlds, int kLen)
{
  const int lane = threadIdx.x & 31;
  const int half = lane >> 4;        // 0: K chunks {0..7,16..23}; 1: {8..15,24..31}
  const int r    = lane & 15;        // row within the 16-row tile
  const float* rp = Wrow0 + (size_t)r * (size_t)ldw;
  v8f acc = {};
  for (int k0 = 0; k0 < kLen; k0 += 32) {
    const int j0 = k0 + half * 8;
    // A: 16 fp32 weights -> f16 (two contiguous 32B chunks per lane; together
    // lanes L and L+16 cover a full contiguous 128B of each row).
    float4 a0 = *(const float4*)(rp + j0);
    float4 a1 = *(const float4*)(rp + j0 + 4);
    float4 a2 = *(const float4*)(rp + j0 + 16);
    float4 a3 = *(const float4*)(rp + j0 + 20);
    v16h A;
    A[0]=(_Float16)a0.x;  A[1]=(_Float16)a0.y;  A[2]=(_Float16)a0.z;  A[3]=(_Float16)a0.w;
    A[4]=(_Float16)a1.x;  A[5]=(_Float16)a1.y;  A[6]=(_Float16)a1.z;  A[7]=(_Float16)a1.w;
    A[8]=(_Float16)a2.x;  A[9]=(_Float16)a2.y;  A[10]=(_Float16)a2.z; A[11]=(_Float16)a2.w;
    A[12]=(_Float16)a3.x; A[13]=(_Float16)a3.y; A[14]=(_Float16)a3.z; A[15]=(_Float16)a3.w;
    // B: broadcast vector (f16 in LDS), same K slot pattern as A for this lane.
    v8h b0 = *(const v8h*)(xlds + j0);
    v8h b1 = *(const v8h*)(xlds + j0 + 16);
    v16h B;
    #pragma unroll
    for (int i = 0; i < 8; ++i) { B[i] = b0[i]; B[8 + i] = b1[i]; }
    acc = __builtin_amdgcn_wmma_f32_16x16x32_f16(
        /*neg_a=*/false, A, /*neg_b=*/false, B,
        /*c_mod=*/(short)0, acc, /*reuse_a=*/false, /*reuse_b=*/false);
  }
  return acc;
}

// ---- Kernel 0: build x (relu + L2 normalize), store f16 copy ----------------
__global__ __launch_bounds__(256)
void k_prep(const float* __restrict__ blat_e, const float* __restrict__ bfb_e,
            const float* __restrict__ tau,    const float* __restrict__ thr,
            _Float16* __restrict__ x16)
{
  const int c = blockIdx.x;                    // 8 blocks x 256 threads
  const float a   = expf(-1.0f / tau[c]);
  const float t   = thr[c];
  const float oma = 1.0f - a;
  float pre[4]; float ss = 0.0f;
  #pragma unroll
  for (int j = 0; j < 4; ++j) {
    const int h = threadIdx.x + j * 256;
    float v = oma * (blat_e[c * (L_ * H_) + h] + bfb_e[c * (3 * H_) + h]) - t;
    v = fmaxf(v, 0.0f);
    pre[j] = v; ss += v * v;
  }
  __shared__ float red[256];
  red[threadIdx.x] = ss; __syncthreads();
  for (int off = 128; off > 0; off >>= 1) {
    if (threadIdx.x < off) red[threadIdx.x] += red[threadIdx.x + off];
    __syncthreads();
  }
  const float inv = 1.0f / (sqrtf(red[0]) + 1e-8f);
  #pragma unroll
  for (int j = 0; j < 4; ++j) {
    const int h = threadIdx.x + j * 256;
    x16[c * H_ + h] = (_Float16)(pre[j] * inv);
  }
}

// ---- Kernel 1: out[c] = relu(Wexc[c,0] @ x[c] + bexc[c,0]) ------------------
__global__ __launch_bounds__(256)
void k_exc(const float* __restrict__ Wexc, const float* __restrict__ bexc,
           const _Float16* __restrict__ x16g,
           float* __restrict__ out_f32, _Float16* __restrict__ out_f16)
{
  __shared__ _Float16 xl[H_];
  const int c = blockIdx.x >> 3;               // 64 blocks: 8 per circuit
  for (int i = threadIdx.x * 8; i < H_; i += 256 * 8)
    *(v8h*)(xl + i) = *(const v8h*)(x16g + c * H_ + i);
  __syncthreads();

  const int wave = (blockIdx.x & 7) * 8 + (threadIdx.x >> 5);  // 0..63 in c
  const int m0 = wave * 16;
  const float* W0 = Wexc + (size_t)c * (L_ * H_ * H_) + (size_t)m0 * H_;
  v8f acc = wmma_matvec_rows16(W0, H_, xl, H_);

  const int lane = threadIdx.x & 31;
  if ((lane & 15) == 0) {                       // lanes 0 and 16: D column N=0
    const int half = lane >> 4;
    #pragma unroll
    for (int rr = 0; rr < 8; ++rr) {
      const int m = m0 + half * 8 + rr;
      float v = acc[rr] + bexc[c * (L_ * H_) + m];
      v = fmaxf(v, 0.0f);
      out_f32[c * H_ + m] = v;
      out_f16[c * H_ + m] = (_Float16)v;
    }
  }
}

// ---- Kernel 2: partials of Wg1 (1024 x 8192) @ out, K split 8 ways ----------
__global__ __launch_bounds__(256)
void k_g1(const float* __restrict__ Wg1, const _Float16* __restrict__ v16g,
          float* __restrict__ partial)
{
  __shared__ _Float16 vl[CH_];
  for (int i = threadIdx.x * 8; i < CH_; i += 256 * 8)
    *(v8h*)(vl + i) = *(const v8h*)(v16g + i);
  __syncthreads();

  const int w    = blockIdx.x * 8 + (threadIdx.x >> 5);  // 0..511
  const int s    = w & 7;                                // K chunk of 1024
  const int m0   = (w >> 3) * 16;                        // row tile
  const float* W0 = Wg1 + (size_t)m0 * CH_ + (size_t)s * H_;
  v8f acc = wmma_matvec_rows16(W0, CH_, vl + s * H_, H_);

  const int lane = threadIdx.x & 31;
  if ((lane & 15) == 0) {
    const int half = lane >> 4;
    #pragma unroll
    for (int rr = 0; rr < 8; ++rr) {
      const int m = m0 + half * 8 + rr;
      partial[s * H_ + m] = acc[rr];            // unique (s,m): deterministic
    }
  }
}

// ---- Kernel 3: h = relu(sum_s partial + bg1), f16 copy ----------------------
__global__ __launch_bounds__(256)
void k_g1_reduce(const float* __restrict__ partial, const float* __restrict__ bg1,
                 _Float16* __restrict__ h16)
{
  const int m = blockIdx.x * 256 + threadIdx.x;  // 4 blocks
  float s = bg1[m];
  #pragma unroll
  for (int j = 0; j < 8; ++j) s += partial[j * H_ + m];
  h16[m] = (_Float16)fmaxf(s, 0.0f);
}

// ---- Kernel 4: final = out * (1 - sigmoid(Wg2 @ h + bg2)) -------------------
__global__ __launch_bounds__(256)
void k_g2(const float* __restrict__ Wg2, const float* __restrict__ bg2,
          const _Float16* __restrict__ h16, const float* __restrict__ out_f32,
          float* __restrict__ fin)
{
  __shared__ _Float16 hl[H_];
  for (int i = threadIdx.x * 8; i < H_; i += 256 * 8)
    *(v8h*)(hl + i) = *(const v8h*)(h16 + i);
  __syncthreads();

  const int w  = blockIdx.x * 8 + (threadIdx.x >> 5);    // 0..511
  const int m0 = w * 16;
  const float* W0 = Wg2 + (size_t)m0 * H_;
  v8f acc = wmma_matvec_rows16(W0, H_, hl, H_);

  const int lane = threadIdx.x & 31;
  if ((lane & 15) == 0) {
    const int half = lane >> 4;
    #pragma unroll
    for (int rr = 0; rr < 8; ++rr) {
      const int m = m0 + half * 8 + rr;
      const float z = acc[rr] + bg2[m];
      const float g = 1.0f / (1.0f + expf(-z));
      fin[m] = out_f32[m] * (1.0f - g);
    }
  }
}

extern "C" void kernel_launch(void* const* d_in, const int* in_sizes, int n_in,
                              void* d_out, int out_size, void* d_ws, size_t ws_size,
                              hipStream_t stream) {
  (void)in_sizes; (void)n_in; (void)out_size; (void)ws_size;
  const float* bfb_e   = (const float*)d_in[8];
  const float* blat_e  = (const float*)d_in[12];
  const float* Wexc    = (const float*)d_in[15];
  const float* bexc    = (const float*)d_in[16];
  const float* tau_exc = (const float*)d_in[19];
  const float* thresh  = (const float*)d_in[21];
  const float* Wg1     = (const float*)d_in[24];
  const float* bg1     = (const float*)d_in[25];
  const float* Wg2     = (const float*)d_in[26];
  const float* bg2     = (const float*)d_in[27];

  char* ws = (char*)d_ws;
  _Float16* x16     = (_Float16*)(ws +      0);   // 16 KB
  float*    out_f32 = (float*)   (ws +  16384);   // 32 KB
  _Float16* out_f16 = (_Float16*)(ws +  49152);   // 16 KB
  float*    partial = (float*)   (ws +  65536);   // 32 KB
  _Float16* h16     = (_Float16*)(ws +  98304);   //  2 KB

  k_prep     <<<C_, 256, 0, stream>>>(blat_e, bfb_e, tau_exc, thresh, x16);
  k_exc      <<<64, 256, 0, stream>>>(Wexc, bexc, x16, out_f32, out_f16);
  k_g1       <<<64, 256, 0, stream>>>(Wg1, out_f16, partial);
  k_g1_reduce<<< 4, 256, 0, stream>>>(partial, bg1, h16);
  k_g2       <<<64, 256, 0, stream>>>(Wg2, bg2, h16, out_f32, (float*)d_out);
}
